// TransformerFeatureExtractor_4458176053365
// MI455X (gfx1250) — compile-verified, matrix operand order checked by reading, not run on hardware
//
#include <hip/hip_runtime.h>

// ---------------------------------------------------------------------------
// Problem constants (B=4, NX=NY=1024, E=512, H=8, DH=64, FF=2048, L=4)
// ---------------------------------------------------------------------------
#define SCALE_F (0.125f)      // DH^-0.5
#define EPS_F   (1e-5f)

typedef __attribute__((ext_vector_type(16))) __bf16 v16bf;
typedef __attribute__((ext_vector_type(8)))  float  v8f;

__device__ __forceinline__ unsigned short f2bf(float f) {
  unsigned u = __float_as_uint(f);
  u += 0x7FFFu + ((u >> 16) & 1u);         // round-to-nearest-even
  return (unsigned short)(u >> 16);
}
__device__ __forceinline__ float bf2f(unsigned short h) {
  return __uint_as_float(((unsigned)h) << 16);
}
// LDS byte offset of a __shared__ object: flat LDS-aperture addr low 32 bits
// (ISA 10.2).
__device__ __forceinline__ unsigned lds_off(const void* p) {
  return (unsigned)(unsigned long long)p;
}

// ---------------------------------------------------------------------------
// f32 -> bf16 conversion (activations)
// ---------------------------------------------------------------------------
__global__ void cvt_bf16_kernel(const float* __restrict__ s,
                                unsigned short* __restrict__ d, int n) {
  int i = blockIdx.x * blockDim.x + threadIdx.x;
  if (i < n) d[i] = f2bf(s[i]);
}

// ---------------------------------------------------------------------------
// Weight f32 (K x N) -> bf16 transposed (N x K), 32x32 LDS tiles, 256 thr.
// grid = (K/32, N/32)
// ---------------------------------------------------------------------------
__global__ __launch_bounds__(256)
void cvt_w_transpose_kernel(const float* __restrict__ s,
                            unsigned short* __restrict__ d, int Kd, int Nd) {
  __shared__ float t[32][33];
  int kb = blockIdx.x * 32, nb = blockIdx.y * 32;
  int tx = threadIdx.x & 31, ty = threadIdx.x >> 5;   // ty 0..7
#pragma unroll
  for (int i = ty; i < 32; i += 8)
    t[i][tx] = s[(long long)(kb + i) * Nd + nb + tx];
  __syncthreads();
#pragma unroll
  for (int i = ty; i < 32; i += 8)
    d[(long long)(nb + i) * Kd + kb + tx] = f2bf(t[tx][i]);
}

// ---------------------------------------------------------------------------
// V (4096 x 512 bf16) -> per-head transposed vT (32 batches of 64 x 1024).
// grid = (1024/32, 64/32, 32), 256 thr.
// ---------------------------------------------------------------------------
__global__ __launch_bounds__(256)
void v_transpose_kernel(const unsigned short* __restrict__ v,
                        unsigned short* __restrict__ vt) {
  __shared__ unsigned short t[32][33];
  int mb = blockIdx.x * 32, db = blockIdx.y * 32, z = blockIdx.z;
  int b = z >> 3, h = z & 7;
  int tx = threadIdx.x & 31, ty = threadIdx.x >> 5;
#pragma unroll
  for (int i = ty; i < 32; i += 8)
    t[i][tx] = v[(long long)(b * 1024 + mb + i) * 512 + h * 64 + db + tx];
  __syncthreads();
#pragma unroll
  for (int i = ty; i < 32; i += 8)
    vt[(long long)z * 65536 + (long long)(db + i) * 1024 + mb + tx] = t[tx][i];
}

// ---------------------------------------------------------------------------
// Fragment read from an LDS tile (row stride 40 bf16, K contiguous).
// ISA 7.12.2 16-bit layout: lanes 0-15 hold K 0-7 / 16-23, lanes 16-31 hold
// K 8-15 / 24-31, consecutive-K pairs packed per VGPR.
// ---------------------------------------------------------------------------
__device__ __forceinline__ v16bf read_frag(const unsigned short* base,
                                           int row0, int lane) {
  int g = (lane >> 4) & 1;
  int r = lane & 15;
  const unsigned short* p = base + (row0 + r) * 40;
  union { unsigned u[8]; v16bf v; } f;
#pragma unroll
  for (int i = 0; i < 4; ++i) {
    f.u[i]     = *(const unsigned*)(p + g * 8 + 2 * i);
    f.u[i + 4] = *(const unsigned*)(p + 16 + g * 8 + 2 * i);
  }
  return f.v;
}

// ---------------------------------------------------------------------------
// Batched bf16 NT GEMM: C = A (M x K, row-major) @ B^T (B is N x K row-major)
// + bias (+GELU); f32 and/or bf16 outputs.
// Block tile: 128 x (64*BN_WAVES); 4 x BN_WAVES waves; wave tile 32 x 64
// (2 A-frags x 4 B-frags -> 8 WMMAs / 32-K step).
// Global -> LDS staging uses CDNA5 async loads (ASYNCcnt) with double buffer.
// ---------------------------------------------------------------------------
template <int BN_WAVES>
__global__ __launch_bounds__(128 * BN_WAVES)
void gemm_bf16_nt_kernel(const unsigned short* __restrict__ A, int lda, long long sA,
                         const unsigned short* __restrict__ Bm, int ldb, long long sB,
                         const float* __restrict__ bias,
                         float* __restrict__ Cf, unsigned short* __restrict__ Cb,
                         int ldc, long long csOuter, long long csInner, int cInnerN,
                         int K, int gelu) {
  constexpr int BM = 128;
  constexpr int BN = 64 * BN_WAVES;
  constexpr int T  = 128 * BN_WAVES;
  __shared__ unsigned short As[2][BM][40];
  __shared__ unsigned short Bs[2][BN][40];

  const int tid  = threadIdx.x;
  const int lane = tid & 31;
  const int w    = tid >> 5;
  const int wr   = w & 3;          // wave row   (0..3)
  const int wc   = w >> 2;         // wave col   (0..BN_WAVES-1)
  const int m0   = blockIdx.y * BM;
  const int n0   = blockIdx.x * BN;
  const int z    = blockIdx.z;

  const unsigned short* Ab = A  + (long long)z * sA + (long long)m0 * lda;
  const unsigned short* Bb = Bm + (long long)z * sB + (long long)n0 * ldb;
  const unsigned aLds = lds_off(&As[0][0][0]);
  const unsigned bLds = lds_off(&Bs[0][0][0]);
  constexpr unsigned aBufB = (unsigned)(BM * 40 * 2);
  constexpr unsigned bBufB = (unsigned)(BN * 40 * 2);

  // Constexpr trip counts -> no dynamic bounds checks / EXEC churn in loop.
  auto issue_tile = [&](int kb, int buf) {
    constexpr int ACH = (BM * 4) / T;        // 16B chunks of A per thread
#pragma unroll
    for (int c = 0; c < ACH; ++c) {
      int i = tid + c * T;
      int row = i >> 2, seg = i & 3;
      const unsigned short* gp = Ab + (long long)row * lda + kb + seg * 8;
      unsigned l = aLds + (unsigned)buf * aBufB + (unsigned)(row * 80 + seg * 16);
      asm volatile("global_load_async_to_lds_b128 %0, %1, off"
                   :: "v"(l), "v"(gp) : "memory");
    }
    constexpr int BCH = (BN * 4) / T;        // 16B chunks of B per thread
#pragma unroll
    for (int c = 0; c < BCH; ++c) {
      int i = tid + c * T;
      int row = i >> 2, seg = i & 3;
      const unsigned short* gp = Bb + (long long)row * ldb + kb + seg * 8;
      unsigned l = bLds + (unsigned)buf * bBufB + (unsigned)(row * 80 + seg * 16);
      asm volatile("global_load_async_to_lds_b128 %0, %1, off"
                   :: "v"(l), "v"(gp) : "memory");
    }
  };

  v8f vz = {0, 0, 0, 0, 0, 0, 0, 0};
  v8f acc[2][4];
#pragma unroll
  for (int tm = 0; tm < 2; ++tm)
#pragma unroll
    for (int tn = 0; tn < 4; ++tn) acc[tm][tn] = vz;

  issue_tile(0, 0);
  asm volatile("s_wait_asynccnt 0x0" ::: "memory");
  __syncthreads();

  const int steps = K >> 5;
  for (int s = 0; s < steps; ++s) {
    const int cur = s & 1;
    if (s + 1 < steps) issue_tile((s + 1) * 32, cur ^ 1);

    const unsigned short* aT = &As[cur][0][0];
    const unsigned short* bT = &Bs[cur][0][0];
    v16bf a0 = read_frag(aT, wr * 32,      lane);
    v16bf a1 = read_frag(aT, wr * 32 + 16, lane);
    v16bf b0 = read_frag(bT, wc * 64,      lane);
    v16bf b1 = read_frag(bT, wc * 64 + 16, lane);
    v16bf b2 = read_frag(bT, wc * 64 + 32, lane);
    v16bf b3 = read_frag(bT, wc * 64 + 48, lane);

    acc[0][0] = __builtin_amdgcn_wmma_f32_16x16x32_bf16(false, a0, false, b0, (short)0, acc[0][0], false, false);
    acc[0][1] = __builtin_amdgcn_wmma_f32_16x16x32_bf16(false, a0, false, b1, (short)0, acc[0][1], false, false);
    acc[0][2] = __builtin_amdgcn_wmma_f32_16x16x32_bf16(false, a0, false, b2, (short)0, acc[0][2], false, false);
    acc[0][3] = __builtin_amdgcn_wmma_f32_16x16x32_bf16(false, a0, false, b3, (short)0, acc[0][3], false, false);
    acc[1][0] = __builtin_amdgcn_wmma_f32_16x16x32_bf16(false, a1, false, b0, (short)0, acc[1][0], false, false);
    acc[1][1] = __builtin_amdgcn_wmma_f32_16x16x32_bf16(false, a1, false, b1, (short)0, acc[1][1], false, false);
    acc[1][2] = __builtin_amdgcn_wmma_f32_16x16x32_bf16(false, a1, false, b2, (short)0, acc[1][2], false, false);
    acc[1][3] = __builtin_amdgcn_wmma_f32_16x16x32_bf16(false, a1, false, b3, (short)0, acc[1][3], false, false);

    asm volatile("s_wait_asynccnt 0x0" ::: "memory");
    __syncthreads();
  }

  // Epilogue. D layout: VGPR r -> M = r + 8*(lane/16), N = lane%16.
  long long cbase = (long long)(z / cInnerN) * csOuter +
                    (long long)(z % cInnerN) * csInner;
  int g  = (lane >> 4) & 1;
  int nn = lane & 15;
#pragma unroll
  for (int tm = 0; tm < 2; ++tm) {
#pragma unroll
    for (int tn = 0; tn < 4; ++tn) {
      int nglob = n0 + wc * 64 + tn * 16 + nn;
      float bvv = bias ? bias[nglob] : 0.0f;
      int mloc = m0 + wr * 32 + tm * 16 + g * 8;
#pragma unroll
      for (int r = 0; r < 8; ++r) {
        float v = acc[tm][tn][r] + bvv;
        if (gelu) v = 0.5f * v * (1.0f + erff(v * 0.70710678118654752f));
        long long idx = cbase + (long long)(mloc + r) * ldc + nglob;
        if (Cf) Cf[idx] = v;
        if (Cb) Cb[idx] = f2bf(v);
      }
    }
  }
}

// ---------------------------------------------------------------------------
// Softmax over NY=1024 with mask + residual-attention accumulate (bf16 state).
// ---------------------------------------------------------------------------
__global__ __launch_bounds__(256)
void softmax_attn_kernel(const float* __restrict__ scores,
                         const int* __restrict__ xm, const int* __restrict__ ym,
                         unsigned short* __restrict__ attn, int first) {
  __shared__ float red[256];
  int row = blockIdx.x;          // (b*8+h)*1024 + n'
  int bh  = row >> 10;
  int b   = bh >> 3;
  int n   = row & 1023;
  int tid = threadIdx.x;
  const float* srow = scores + (long long)row * 1024;
  unsigned short* arow = attn + (long long)row * 1024;
  int rowmask = xm[b * 1024 + n];

  float ls[4];
#pragma unroll
  for (int j = 0; j < 4; ++j) {
    int c = j * 256 + tid;
    float s = srow[c] * SCALE_F;
    if (rowmask == 0 || ym[b * 1024 + c] == 0) s = -1e9f;
    ls[j] = s;
  }
  float m = fmaxf(fmaxf(ls[0], ls[1]), fmaxf(ls[2], ls[3]));
  red[tid] = m; __syncthreads();
  for (int s = 128; s > 0; s >>= 1) {
    if (tid < s) red[tid] = fmaxf(red[tid], red[tid + s]);
    __syncthreads();
  }
  m = red[0]; __syncthreads();

  float e[4], sum = 0.0f;
#pragma unroll
  for (int j = 0; j < 4; ++j) { e[j] = __expf(ls[j] - m); sum += e[j]; }
  red[tid] = sum; __syncthreads();
  for (int s = 128; s > 0; s >>= 1) {
    if (tid < s) red[tid] += red[tid + s];
    __syncthreads();
  }
  float inv = 1.0f / red[0];
#pragma unroll
  for (int j = 0; j < 4; ++j) {
    int c = j * 256 + tid;
    float p = e[j] * inv;
    if (!first) p += bf2f(arow[c]);
    arow[c] = f2bf(p);
  }
}

// ---------------------------------------------------------------------------
// out = LayerNorm(X + Y) * g + b ; optional f32 and bf16 outputs. Row = 512.
// ---------------------------------------------------------------------------
__global__ __launch_bounds__(256)
void add_ln_kernel(const float* __restrict__ X, const float* __restrict__ Y,
                   const float* __restrict__ g, const float* __restrict__ bta,
                   float* __restrict__ outF, unsigned short* __restrict__ outB) {
  __shared__ float red[256];
  int row = blockIdx.x, tid = threadIdx.x;
  const float* xr = X + (long long)row * 512;
  const float* yr = Y + (long long)row * 512;
  float v0 = xr[tid] + yr[tid];
  float v1 = xr[tid + 256] + yr[tid + 256];
  red[tid] = v0 + v1; __syncthreads();
  for (int s = 128; s > 0; s >>= 1) {
    if (tid < s) red[tid] += red[tid + s];
    __syncthreads();
  }
  float mu = red[0] * (1.0f / 512.0f); __syncthreads();
  float d0 = v0 - mu, d1 = v1 - mu;
  red[tid] = d0 * d0 + d1 * d1; __syncthreads();
  for (int s = 128; s > 0; s >>= 1) {
    if (tid < s) red[tid] += red[tid + s];
    __syncthreads();
  }
  float rs = rsqrtf(red[0] * (1.0f / 512.0f) + EPS_F);
  float o0 = d0 * rs * g[tid] + bta[tid];
  float o1 = d1 * rs * g[tid + 256] + bta[tid + 256];
  long long base = (long long)row * 512;
  if (outF) { outF[base + tid] = o0; outF[base + tid + 256] = o1; }
  if (outB) { outB[base + tid] = f2bf(o0); outB[base + tid + 256] = f2bf(o1); }
}

// ---------------------------------------------------------------------------
// Host-side orchestration
// ---------------------------------------------------------------------------
extern "C" void kernel_launch(void* const* d_in, const int* in_sizes, int n_in,
                              void* d_out, int out_size, void* d_ws, size_t ws_size,
                              hipStream_t stream) {
  (void)in_sizes; (void)n_in; (void)out_size; (void)ws_size;
  const float* x   = (const float*)d_in[0];
  const float* y   = (const float*)d_in[1];
  const int*   xm  = (const int*)d_in[2];
  const int*   ym  = (const int*)d_in[3];
  const float* Wq  = (const float*)d_in[4];
  const float* bq  = (const float*)d_in[5];
  const float* Wk  = (const float*)d_in[6];
  const float* bk  = (const float*)d_in[7];
  const float* Wv  = (const float*)d_in[8];
  const float* bv  = (const float*)d_in[9];
  const float* Wo  = (const float*)d_in[10];
  const float* bo  = (const float*)d_in[11];
  const float* g1  = (const float*)d_in[12];
  const float* be1 = (const float*)d_in[13];
  const float* g2  = (const float*)d_in[14];
  const float* be2 = (const float*)d_in[15];
  const float* W1  = (const float*)d_in[16];
  const float* bf1 = (const float*)d_in[17];
  const float* W2  = (const float*)d_in[18];
  const float* bf2 = (const float*)d_in[19];

  unsigned char* ws = (unsigned char*)d_ws;
  size_t off = 0;
  auto alloc = [&](size_t bytes) -> void* {
    void* p = ws + off;
    off += (bytes + 255) & ~(size_t)255;
    return p;
  };

  const size_t TOK = 4096;            // B*NX rows
  unsigned short* xb   = (unsigned short*)alloc(TOK * 512 * 2);
  unsigned short* yb   = (unsigned short*)alloc(TOK * 512 * 2);
  unsigned short* qb   = (unsigned short*)alloc(TOK * 512 * 2);
  unsigned short* kb_  = (unsigned short*)alloc(TOK * 512 * 2);
  unsigned short* vb_  = (unsigned short*)alloc(TOK * 512 * 2);
  unsigned short* vT   = (unsigned short*)alloc(TOK * 512 * 2);  // 32 x (64x1024)
  unsigned short* ob   = (unsigned short*)alloc(TOK * 512 * 2);
  unsigned short* hb   = (unsigned short*)alloc(TOK * 512 * 2);
  unsigned short* f1b  = (unsigned short*)alloc(TOK * 2048 * 2);
  float* scores = (float*)alloc((size_t)32 * 1024 * 1024 * 4);
  unsigned short* attn = (unsigned short*)alloc((size_t)32 * 1024 * 1024 * 2);
  float* oproj = (float*)alloc(TOK * 512 * 4);
  float* hf    = (float*)alloc(TOK * 512 * 4);
  float* ff2f  = (float*)alloc(TOK * 512 * 4);
  unsigned short* wqT = (unsigned short*)alloc(512 * 512 * 2);
  unsigned short* wkT = (unsigned short*)alloc(512 * 512 * 2);
  unsigned short* wvT = (unsigned short*)alloc(512 * 512 * 2);
  unsigned short* woT = (unsigned short*)alloc(512 * 512 * 2);
  unsigned short* w1T = (unsigned short*)alloc(512 * 2048 * 2);
  unsigned short* w2T = (unsigned short*)alloc(2048 * 512 * 2);

  auto cvt = [&](const float* s, unsigned short* d, int n) {
    cvt_bf16_kernel<<<(n + 255) / 256, 256, 0, stream>>>(s, d, n);
  };
  auto wtr = [&](const float* s, unsigned short* d, int Kd, int Nd) {
    cvt_w_transpose_kernel<<<dim3(Kd / 32, Nd / 32), 256, 0, stream>>>(s, d, Kd, Nd);
  };
  // All GEMMs are NT: B supplied as (N x K) row-major.
  auto gemm = [&](int bnw, const unsigned short* A, int lda, long long sA,
                  const unsigned short* Bmat, int ldb, long long sB,
                  const float* bias, float* Cf, unsigned short* Cb,
                  int ldc, long long csOuter, long long csInner, int cInnerN,
                  int M, int N, int K, int nbatch, int gelu) {
    dim3 grid(N / (64 * bnw), M / 128, nbatch), blk(128 * bnw);
    if (bnw == 2)
      gemm_bf16_nt_kernel<2><<<grid, blk, 0, stream>>>(A, lda, sA, Bmat, ldb, sB,
          bias, Cf, Cb, ldc, csOuter, csInner, cInnerN, K, gelu);
    else
      gemm_bf16_nt_kernel<1><<<grid, blk, 0, stream>>>(A, lda, sA, Bmat, ldb, sB,
          bias, Cf, Cb, ldc, csOuter, csInner, cInnerN, K, gelu);
  };

  cvt(x, xb, 4096 * 512);
  cvt(y, yb, 4096 * 512);

  // Layers 0..3: only QK-proj + scores + softmax/attn-residual are live
  // (x/y never update; only the attention residual chains).
  for (int l = 0; l < 4; ++l) {
    wtr(Wq + (size_t)l * 262144, wqT, 512, 512);
    wtr(Wk + (size_t)l * 262144, wkT, 512, 512);
    gemm(2, xb, 512, 0, wqT, 512, 0, bq + l * 512, nullptr, qb,
         512, 0, 0, 1, 4096, 512, 512, 1, 0);
    gemm(2, yb, 512, 0, wkT, 512, 0, bk + l * 512, nullptr, kb_,
         512, 0, 0, 1, 4096, 512, 512, 1, 0);
    // scores[bh] = Qh(1024x64) @ Kh(1024x64)^T  (reshape is a flat view)
    gemm(2, qb, 64, 65536, kb_, 64, 65536, nullptr, scores, nullptr,
         1024, 1048576, 0, 1, 1024, 1024, 64, 32, 0);
    softmax_attn_kernel<<<32 * 1024, 256, 0, stream>>>(scores, xm, ym, attn,
                                                       l == 0 ? 1 : 0);
  }

  // Final layer only: V, A@V, Wo, LN1, FFN(+GELU), LN2 -> d_out.
  wtr(Wv + 3 * 262144, wvT, 512, 512);
  wtr(Wo + 3 * 262144, woT, 512, 512);
  wtr(W1 + 3 * 1048576, w1T, 512, 2048);
  wtr(W2 + 3 * 1048576, w2T, 2048, 512);

  gemm(2, yb, 512, 0, wvT, 512, 0, bv + 3 * 512, nullptr, vb_,
       512, 0, 0, 1, 4096, 512, 512, 1, 0);
  v_transpose_kernel<<<dim3(32, 2, 32), 256, 0, stream>>>(vb_, vT);
  // o[bh] = attn(1024x1024) @ Vh ; C tile for (b,h) lands at row n',
  // col h*64 of the merged (4096 x 512) output (the transpose+reshape).
  gemm(1, attn, 1024, 1048576, vT, 1024, 65536, nullptr, nullptr, ob,
       512, 524288, 64, 8, 1024, 64, 1024, 32, 0);
  gemm(2, ob, 512, 0, woT, 512, 0, bo + 3 * 512, oproj, nullptr,
       512, 0, 0, 1, 4096, 512, 512, 1, 0);
  add_ln_kernel<<<4096, 256, 0, stream>>>(x, oproj, g1 + 3 * 512, be1 + 3 * 512,
                                          hf, hb);
  gemm(2, hb, 512, 0, w1T, 512, 0, bf1 + 3 * 2048, nullptr, f1b,
       2048, 0, 0, 1, 4096, 2048, 512, 1, 1 /*gelu*/);
  gemm(2, f1b, 2048, 0, w2T, 2048, 0, bf2 + 3 * 512, ff2f, nullptr,
       512, 0, 0, 1, 4096, 512, 2048, 1, 0);
  add_ln_kernel<<<4096, 256, 0, stream>>>(hf, ff2f, g2 + 3 * 512, be2 + 3 * 512,
                                          (float*)d_out, nullptr);
}
